// NetVLADLayer_46205258170690
// MI455X (gfx1250) — matrix operand order repaired and checked by reading, not compiled, not run on hardware
//
#include <hip/hip_runtime.h>
#include <hip/hip_bf16.h>

// ---------------------------------------------------------------------------
// NetVLAD on MI455X (gfx1250):
//   - stage f32 descriptors -> bf16 LDS tiles (read HBM once: ~134 MB floor)
//   - GEMM1 (scores) and GEMM2 (agg) both via v_wmma_f32_16x16x32_bf16
//   - fragment loads batched ahead of WMMA bursts so s_wait_dscnt does not
//     serialize every matrix op against LDS latency
// Memory roofline: 134MB / 23.3TB/s ~ 5.8us; bf16 WMMA keeps the 8.6 GFLOP
// of matmul at/below that roof (f32 WMMA 16x16x4 would be ~8x compute-bound).
// ---------------------------------------------------------------------------

typedef __attribute__((ext_vector_type(16))) __bf16 v16bf;
typedef __attribute__((ext_vector_type(8)))  __bf16 v8bf;
typedef __attribute__((ext_vector_type(2)))  __bf16 v2bf;
typedef __attribute__((ext_vector_type(8)))  float  v8f;

static constexpr int B_ = 16, D_ = 512, K_ = 64, N_ = 4096;
static constexpr int NT       = 64;            // n-tile width
static constexpr int CHUNKS   = 8;             // N-chunks per batch
static constexpr int CHUNK_N  = N_ / CHUNKS;   // 512
static constexpr int TILES    = CHUNK_N / NT;  // 8
static constexpr int THREADS  = 512;           // 16 wave32

// LDS row strides (elements) — padded for bank spread + 16B alignment
static constexpr int WS  = 520;  // W rows (bf16)
static constexpr int DAS = 72;   // desc d-major rows (bf16)
static constexpr int DBS = 520;  // desc n-major rows (bf16)
static constexpr int RS  = 68;   // raw scores rows (f32)
static constexpr int SS  = 72;   // bf16 scores rows

static constexpr size_t LDS_BYTES =
    (size_t)(64 * WS + 512 * DAS + 64 * DBS + 64 * SS) * 2 +
    (size_t)(64 * RS + 64 + 64) * 4;   // = 233,984 B  (< 320 KB WGP LDS)

__device__ __forceinline__ v16bf join16(v8bf a, v8bf b) {
  v16bf r;
#pragma unroll
  for (int i = 0; i < 8; ++i) { r[i] = a[i]; r[8 + i] = b[i]; }
  return r;
}

__global__ void __launch_bounds__(THREADS)
netvlad_main(const float* __restrict__ desc, const float* __restrict__ Wg,
             const float* __restrict__ bg, float* __restrict__ agg,
             float* __restrict__ ssum_g) {
  extern __shared__ char smem[];
  __bf16* Wl   = (__bf16*)smem;            // [64][520]  W as bf16
  __bf16* dA   = Wl + 64 * WS;             // [512][72]  desc tile, d-major
  __bf16* dB   = dA + 512 * DAS;           // [64][520]  desc tile, n-major
  float*  raw  = (float*)(dB + 64 * DBS);  // [64][68]   raw logits
  __bf16* sco  = (__bf16*)(raw + 64 * RS); // [64][72]   softmax probs (k-major)
  float*  ssum = (float*)(sco + 64 * SS);  // [64]       per-chunk ssum accum
  float*  bias = ssum + 64;                // [64]

  const int t    = threadIdx.x;
  const int wave = t >> 5;
  const int lane = t & 31;
  const int b    = blockIdx.x / CHUNKS;
  const int n0c  = (blockIdx.x % CHUNKS) * CHUNK_N;

  // ---- stage W (f32 -> bf16) once per workgroup --------------------------
  {
    const float4* W4 = (const float4*)Wg;
#pragma unroll
    for (int i = 0; i < 16; ++i) {
      int f = t + i * THREADS;            // 8192 float4 total
      float4 v = W4[f];
      int k = f >> 7;                     // /128 float4 per row
      int d = (f & 127) << 2;
      __bf16* p = Wl + k * WS + d;
      p[0] = (__bf16)v.x; p[1] = (__bf16)v.y;
      p[2] = (__bf16)v.z; p[3] = (__bf16)v.w;
    }
    if (t < K_) { bias[t] = bg[t]; ssum[t] = 0.f; }
  }

  // persistent GEMM2 accumulators: 8 tiles x 8 VGPRs per wave
  v8f acc2[8];
#pragma unroll
  for (int j = 0; j < 8; ++j)
#pragma unroll
    for (int r = 0; r < 8; ++r) acc2[j][r] = 0.f;

  const int kt  = wave & 3;   // which 16-cluster block
  const int dtb = wave >> 2;  // GEMM1: n-tile block / GEMM2: d-tile group

  for (int it = 0; it < TILES; ++it) {
    const int n0 = n0c + it * NT;
    const float* dbase = desc + ((size_t)b * D_) * N_ + n0;

    __syncthreads();  // protect dA/dB from previous iteration's GEMM2 readers

    // ---- stage descriptor tile, f32 -> bf16, two layouts -----------------
    // process d-row PAIRS so transposed stores are packed 2xbf16 (b32)
#pragma unroll
    for (int i = 0; i < 8; ++i) {
      int p  = t + i * THREADS;           // 4096 pairs of float4
      int d  = (p >> 4) << 1;             // even d
      int nl = (p & 15) << 2;
      const float* g0 = dbase + (size_t)d * N_ + nl;
      float4 v0 = *(const float4*)g0;
      float4 v1 = *(const float4*)(g0 + N_);
      __bf16* pa0 = dA + d * DAS + nl;    // d-major (GEMM2 A operand)
      __bf16* pa1 = pa0 + DAS;
      pa0[0] = (__bf16)v0.x; pa0[1] = (__bf16)v0.y;
      pa0[2] = (__bf16)v0.z; pa0[3] = (__bf16)v0.w;
      pa1[0] = (__bf16)v1.x; pa1[1] = (__bf16)v1.y;
      pa1[2] = (__bf16)v1.z; pa1[3] = (__bf16)v1.w;
      // n-major (GEMM1 B operand): packed (d, d+1) pair per column
      *(v2bf*)(dB + (nl + 0) * DBS + d) = v2bf{(__bf16)v0.x, (__bf16)v1.x};
      *(v2bf*)(dB + (nl + 1) * DBS + d) = v2bf{(__bf16)v0.y, (__bf16)v1.y};
      *(v2bf*)(dB + (nl + 2) * DBS + d) = v2bf{(__bf16)v0.z, (__bf16)v1.z};
      *(v2bf*)(dB + (nl + 3) * DBS + d) = v2bf{(__bf16)v0.w, (__bf16)v1.w};
    }
    if (it + 1 < TILES)                   // gfx1250 global_prefetch_b8
      __builtin_prefetch(dbase + (size_t)t * N_ + NT, 0, 0);
    __syncthreads();

    // ---- GEMM1: scores[64,64] = W[64,512] x desc[512,64] -----------------
    // batch 4 d-steps of fragment loads, then 4 back-to-back WMMAs
    {
      const int mrow = kt * 16 + (lane & 15);   // cluster row
      const int nrow = dtb * 16 + (lane & 15);  // n column
      const int aoff = (lane < 16) ? 0 : 8;
      const int boff = (lane < 16) ? 0 : 8;
      const __bf16* rowA = Wl + mrow * WS + aoff;
      const __bf16* rowB = dB + nrow * DBS + boff;
      v8f acc;
#pragma unroll
      for (int r = 0; r < 8; ++r) acc[r] = 0.f;
#pragma unroll
      for (int d0 = 0; d0 < D_; d0 += 128) {
        v16bf Af[4], Bf[4];
#pragma unroll
        for (int s = 0; s < 4; ++s) {
          const v8bf* pA = (const v8bf*)(rowA + d0 + 32 * s);
          const v8bf* pB = (const v8bf*)(rowB + d0 + 32 * s);
          Af[s] = join16(pA[0], pA[2]);   // K {0..7,16..23} (+8 hi lanes)
          Bf[s] = join16(pB[0], pB[2]);   // K {0..7,16..23} (+8 hi lanes)
        }
#pragma unroll
        for (int s = 0; s < 4; ++s)
          acc = __builtin_amdgcn_wmma_f32_16x16x32_bf16(
              false, Af[s], false, Bf[s], (short)0, acc, false, false);
      }
      const int mb  = kt * 16 + ((lane < 16) ? 0 : 8);
      const int col = dtb * 16 + (lane & 15);
#pragma unroll
      for (int r = 0; r < 8; ++r) raw[(mb + r) * RS + col] = acc[r];
    }
    __syncthreads();

    // ---- softmax over K=64 per column ------------------------------------
    if (t < NT) {
      float mx = -3.4e38f;
#pragma unroll 4
      for (int k = 0; k < K_; ++k) {
        float v = raw[k * RS + t] + bias[k];
        raw[k * RS + t] = v;
        mx = fmaxf(mx, v);
      }
      float sum = 0.f;
#pragma unroll 4
      for (int k = 0; k < K_; ++k) {
        float e = __expf(raw[k * RS + t] - mx);
        raw[k * RS + t] = e;
        sum += e;
      }
      float inv = 1.f / sum;
#pragma unroll 4
      for (int k = 0; k < K_; ++k)
        sco[k * SS + t] = (__bf16)(raw[k * RS + t] * inv);
    }
    __syncthreads();

    // ---- ssum[k] += sum_n p[k,n]  (one thread per k, no race) ------------
    if (t < K_) {
      float s = 0.f;
#pragma unroll 4
      for (int n = 0; n < NT; ++n) s += (float)sco[t * SS + n];
      ssum[t] += s;
    }

    // ---- GEMM2: agg[512,64] += desc[512,NT] x scores^T[NT,64] ------------
    // load all 8 A-fragments up front, then 8 back-to-back WMMAs
    {
      const int kcol = kt * 16 + (lane & 15);
      const int ko   = (lane < 16) ? 0 : 8;
      const int nb   = (lane < 16) ? 0 : 8;
#pragma unroll
      for (int nc = 0; nc < NT; nc += 32) {
        const v8bf* pB = (const v8bf*)(sco + kcol * SS + nc + nb);
        v16bf Bf = join16(pB[0], pB[2]);
        v16bf Af[8];
#pragma unroll
        for (int j = 0; j < 8; ++j) {
          const int drow = (dtb * 8 + j) * 16 + (lane & 15);
          const v8bf* pA = (const v8bf*)(dA + drow * DAS + nc + ko);
          Af[j] = join16(pA[0], pA[2]);
        }
#pragma unroll
        for (int j = 0; j < 8; ++j)
          acc2[j] = __builtin_amdgcn_wmma_f32_16x16x32_bf16(
              false, Af[j], false, Bf, (short)0, acc2[j], false, false);
      }
    }
  }
  __syncthreads();

  // ---- flush partial agg with f32 global atomics -------------------------
  {
    const int k   = kt * 16 + (lane & 15);
    const int dof = (lane < 16) ? 0 : 8;
#pragma unroll
    for (int j = 0; j < 8; ++j) {
      const int db = (dtb * 8 + j) * 16 + dof;
#pragma unroll
      for (int r = 0; r < 8; ++r) {
        const int d = db + r;
        unsafeAtomicAdd(&agg[((size_t)b * D_ + d) * K_ + k], acc2[j][r]);
      }
    }
  }
  if (t < K_) unsafeAtomicAdd(&ssum_g[b * K_ + t], ssum[t]);
}

// --------- finalize: vlad = agg - centers*ssum; intra-norm; global norm ----
__global__ void __launch_bounds__(64)
netvlad_finalize(const float* __restrict__ agg, const float* __restrict__ ssum_g,
                 const float* __restrict__ centers, float* __restrict__ out) {
  __shared__ float red[K_];
  const int b = blockIdx.x, k = threadIdx.x;
  const float ssk = ssum_g[b * K_ + k];
  float* ob = out + (size_t)b * D_ * K_;
  const float* ab = agg + (size_t)b * D_ * K_;

  float ss = 0.f;
  for (int d = 0; d < D_; ++d) {
    float v = ab[d * K_ + k] - centers[d * K_ + k] * ssk;
    ss += v * v;
    ob[d * K_ + k] = v;   // coalesced across k
  }
  const float inv1 = 1.f / fmaxf(sqrtf(ss), 1e-12f);

  float gs = 0.f;
  for (int d = 0; d < D_; ++d) {
    float w = ob[d * K_ + k] * inv1;
    ob[d * K_ + k] = w;
    gs += w * w;
  }
  red[k] = gs;
  __syncthreads();
  for (int off = 32; off > 0; off >>= 1) {
    if (k < off) red[k] += red[k + off];
    __syncthreads();
  }
  const float inv2 = 1.f / fmaxf(sqrtf(red[0]), 1e-12f);
  for (int d = 0; d < D_; ++d) ob[d * K_ + k] *= inv2;
}

__global__ void zero_ws(float* p, int n) {
  int i = blockIdx.x * blockDim.x + threadIdx.x;
  if (i < n) p[i] = 0.f;
}

extern "C" void kernel_launch(void* const* d_in, const int* in_sizes, int n_in,
                              void* d_out, int out_size, void* d_ws, size_t ws_size,
                              hipStream_t stream) {
  const float* desc    = (const float*)d_in[0];  // [B, D, N]
  const float* Wg      = (const float*)d_in[1];  // [K, D]
  const float* bg      = (const float*)d_in[2];  // [K]
  const float* centers = (const float*)d_in[3];  // [D, K]
  float* out = (float*)d_out;                    // [B, D*K]

  float* agg  = (float*)d_ws;                    // [B, D, K]  (2 MB)
  float* ssum = agg + (size_t)B_ * D_ * K_;      // [B, K]

  const int zn = B_ * D_ * K_ + B_ * K_;
  zero_ws<<<(zn + 255) / 256, 256, 0, stream>>>(agg, zn);

  netvlad_main<<<B_ * CHUNKS, THREADS, LDS_BYTES, stream>>>(
      desc, Wg, bg, agg, ssum);

  netvlad_finalize<<<B_, K_, 0, stream>>>(agg, ssum, centers, out);
}
